// GAT_43430709297531
// MI455X (gfx1250) — matrix operand order repaired and checked by reading, not compiled, not run on hardware
//
#include <hip/hip_runtime.h>
#include <hip/hip_bf16.h>

#define kN    100000
#define kE    1600000
#define kET   (kE + kN)       // edges + self loops = 1,700,000
#define kIN   512
#define kH1   64              // HEADS*HID = 8*8
#define kHEADS 8
#define kOUT  40

typedef __attribute__((ext_vector_type(16))) _Float16 v16h;
typedef __attribute__((ext_vector_type(8)))  _Float16 v8h;
typedef __attribute__((ext_vector_type(8)))  float    v8f;
typedef __attribute__((ext_vector_type(2)))  float    v2f;

// ---------- monotone float <-> ordered uint (for atomic segment-max) ----------
__device__ __forceinline__ unsigned f2ord(float f) {
  unsigned i = __float_as_uint(f);
  return (i & 0x80000000u) ? ~i : (i | 0x80000000u);
}
__device__ __forceinline__ float ord2f(unsigned u) {
  unsigned i = (u & 0x80000000u) ? (u & 0x7FFFFFFFu) : ~u;
  return __uint_as_float(i);
}

// ---------- utility: fill u32 ----------
__global__ void k_fill_u32(unsigned* p, unsigned v, long n) {
  long i = (long)blockIdx.x * blockDim.x + threadIdx.x;
  if (i < n) p[i] = v;
}

// ---------- convert x -> f16 ----------
__global__ void k_cvt_f16(const float* __restrict__ src, _Float16* __restrict__ dst, long n) {
  long i = (long)blockIdx.x * blockDim.x + threadIdx.x;
  if (i < n) dst[i] = (_Float16)src[i];
}

// ---------- W1 [512,64] -> W1t_f16 [64,512] (column-major for B fragments) ----------
__global__ void k_w1_transpose(const float* __restrict__ W1, _Float16* __restrict__ W1t) {
  int i = blockIdx.x * blockDim.x + threadIdx.x;       // 64*512
  if (i >= kH1 * kIN) return;
  int j = i / kIN;            // output column of W1 (0..63)
  int k = i - j * kIN;        // k (0..511)
  W1t[(size_t)j * kIN + k] = (_Float16)W1[(size_t)k * kH1 + j];
}

// ---------- GEMM1: h1[N,64] = x_h[N,512] @ W1[512,64], f16 WMMA, f32 accum ----------
// one wave per 16x16 tile; 6250 x 4 tiles = 25000 tiles; 8 waves/block -> 3125 blocks
__global__ void k_gemm1(const _Float16* __restrict__ Ah, const _Float16* __restrict__ Bt,
                        float* __restrict__ C) {
  const int lane = threadIdx.x & 31;
  const int wave = threadIdx.x >> 5;
  const int tile = blockIdx.x * 8 + wave;
  if (tile >= 6250 * 4) return;                 // wave-uniform
  const int nt = tile & 3;
  const int mt = tile >> 2;
  const int ll = lane & 15;
  const int hl = lane >> 4;

  const _Float16* __restrict__ arow = Ah + (size_t)(mt * 16 + ll) * kIN;
  const _Float16* __restrict__ brow = Bt + (size_t)(nt * 16 + ll) * kIN;

  union U16 { v16h v; v8h h[2]; };
  v8f acc = {};
  #pragma unroll 4
  for (int k0 = 0; k0 < kIN; k0 += 32) {
    U16 a, b;
    // A 16x32 f16 fragment: 16 consecutive K per lane (K = half*16 .. +15)
    a.v = *reinterpret_cast<const v16h*>(arow + k0 + hl * 16);
    // B 32x16 f16 fragment (W1 pre-transposed): 16 consecutive K per lane
    b.v = *reinterpret_cast<const v16h*>(brow + k0 + hl * 16);
    __builtin_prefetch(arow + k0 + 64, 0, 1);   // global_prefetch next A chunk
    acc = __builtin_amdgcn_wmma_f32_16x16x32_f16(false, a.v, false, b.v,
                                                 (short)0, acc, false, false);
  }
  // C/D layout: VGPR v -> row = v + 8*half, col = lane&15
  float* __restrict__ cout = C + (size_t)(mt * 16 + hl * 8) * kH1 + nt * 16 + ll;
  #pragma unroll
  for (int v = 0; v < 8; v++) cout[(size_t)v * kH1] = acc[v];
}

// ---------- per-node attention logits, layer 1 ----------
__global__ void k_nodevec1(const float* __restrict__ h1, const float* __restrict__ a_src,
                           const float* __restrict__ a_dst,
                           float* __restrict__ as1, float* __restrict__ ad1) {
  int tid = blockIdx.x * blockDim.x + threadIdx.x;   // N*8
  if (tid >= kN * kHEADS) return;
  int n = tid >> 3, h = tid & 7;
  const float* row = h1 + (size_t)n * kH1 + h * 8;
  float s = 0.f, d = 0.f;
  #pragma unroll
  for (int c = 0; c < 8; c++) {
    float hv = row[c];
    s += hv * a_src[h * 8 + c];
    d += hv * a_dst[h * 8 + c];
  }
  as1[tid] = s;
  ad1[tid] = d;
}

// ---------- layer1 edge pass A: leaky-relu logit + segment max ----------
__global__ void k_edge1_a(const int* __restrict__ ei, const float* __restrict__ as1,
                          const float* __restrict__ ad1, float* __restrict__ e1,
                          unsigned* __restrict__ mx1) {
  long e = (long)blockIdx.x * blockDim.x + threadIdx.x;
  if (e >= kET) return;
  int s = (e < kE) ? ei[e] : (int)(e - kE);
  int d = (e < kE) ? ei[kE + e] : (int)(e - kE);
  #pragma unroll
  for (int h = 0; h < kHEADS; h++) {
    float v = as1[(size_t)s * 8 + h] + ad1[(size_t)d * 8 + h];
    v = (v > 0.f) ? v : 0.2f * v;
    e1[e * 8 + h] = v;
    atomicMax(&mx1[(size_t)d * 8 + h], f2ord(v));
  }
}

// ---------- layer1 edge pass B: exp(e - max) + segment sum (in-place ex) ----------
__global__ void k_edge1_b(const int* __restrict__ ei, float* __restrict__ e1,
                          const unsigned* __restrict__ mx1, float* __restrict__ den1) {
  long e = (long)blockIdx.x * blockDim.x + threadIdx.x;
  if (e >= kET) return;
  int d = (e < kE) ? ei[kE + e] : (int)(e - kE);
  #pragma unroll
  for (int h = 0; h < kHEADS; h++) {
    float ex = expf(e1[e * 8 + h] - ord2f(mx1[(size_t)d * 8 + h]));
    e1[e * 8 + h] = ex;
    atomicAdd(&den1[(size_t)d * 8 + h], ex);
  }
}

// ---------- layer1 edge pass C: scatter alpha * h1[src] ----------
__global__ void k_edge1_c(const int* __restrict__ ei, const float* __restrict__ ex1,
                          const float* __restrict__ den1, const float* __restrict__ h1,
                          float* __restrict__ out1) {
  long e = (long)blockIdx.x * blockDim.x + threadIdx.x;
  if (e >= kET) return;
  int s = (e < kE) ? ei[e] : (int)(e - kE);
  int d = (e < kE) ? ei[kE + e] : (int)(e - kE);
  float al[kHEADS];
  #pragma unroll
  for (int h = 0; h < kHEADS; h++)
    al[h] = ex1[e * 8 + h] / den1[(size_t)d * 8 + h];
  const float4* __restrict__ hs = reinterpret_cast<const float4*>(h1 + (size_t)s * kH1);
  float* __restrict__ od = out1 + (size_t)d * kH1;
  #pragma unroll
  for (int q = 0; q < 16; q++) {              // 16 float4 = 64 channels
    float4 m = hs[q];
    float a = al[q >> 1];                     // head = (q*4)/8
    atomicAdd(&od[q * 4 + 0], a * m.x);
    atomicAdd(&od[q * 4 + 1], a * m.y);
    atomicAdd(&od[q * 4 + 2], a * m.z);
    atomicAdd(&od[q * 4 + 3], a * m.w);
  }
}

// ---------- bias + ELU (in place) ----------
__global__ void k_elu(float* __restrict__ h, const float* __restrict__ b1, long n) {
  long i = (long)blockIdx.x * blockDim.x + threadIdx.x;
  if (i >= n) return;
  float v = h[i] + b1[i & (kH1 - 1)];
  h[i] = (v > 0.f) ? v : expm1f(v);
}

// ---------- GEMM2: hh[N,40] = h2[N,64] @ W2[64,40], f32 WMMA 16x16x4 ----------
// tiles: 6250 x 3 (cols padded to 48, guarded) -> 18750 tiles, 8 waves/block
__global__ void k_gemm2(const float* __restrict__ A, const float* __restrict__ B,
                        float* __restrict__ C) {
  const int lane = threadIdx.x & 31;
  const int wave = threadIdx.x >> 5;
  const int tile = blockIdx.x * 8 + wave;
  if (tile >= 6250 * 3) return;                // wave-uniform
  const int mt = tile / 3;
  const int nt = tile - mt * 3;
  const int ll = lane & 15;
  const int hl = lane >> 4;
  const int col = nt * 16 + ll;
  const bool colok = (col < kOUT);

  const float* __restrict__ arow = A + (size_t)(mt * 16 + ll) * kH1;
  v8f acc = {};
  #pragma unroll 4
  for (int k0 = 0; k0 < kH1; k0 += 4) {
    // A 16x4 f32 fragment: reg0 = K(2*half), reg1 = K(2*half+1)
    v2f a = *reinterpret_cast<const v2f*>(arow + k0 + 2 * hl);
    v2f b;
    int kb = k0 + 2 * hl;
    b.x = colok ? B[(size_t)kb * kOUT + col] : 0.f;
    b.y = colok ? B[(size_t)(kb + 1) * kOUT + col] : 0.f;
    acc = __builtin_amdgcn_wmma_f32_16x16x4_f32(false, a, false, b,
                                                (short)0, acc, false, false);
  }
  if (colok) {
    float* __restrict__ cout = C + (size_t)(mt * 16 + hl * 8) * kOUT + col;
    #pragma unroll
    for (int v = 0; v < 8; v++) cout[(size_t)v * kOUT] = acc[v];
  }
}

// ---------- per-node attention logits, layer 2 (1 head, 40 ch) ----------
__global__ void k_nodevec2(const float* __restrict__ hh, const float* __restrict__ a_src,
                           const float* __restrict__ a_dst,
                           float* __restrict__ as2, float* __restrict__ ad2) {
  int n = blockIdx.x * blockDim.x + threadIdx.x;
  if (n >= kN) return;
  const float* row = hh + (size_t)n * kOUT;
  float s = 0.f, d = 0.f;
  #pragma unroll 8
  for (int j = 0; j < kOUT; j++) {
    float hv = row[j];
    s += hv * a_src[j];
    d += hv * a_dst[j];
  }
  as2[n] = s;
  ad2[n] = d;
}

__global__ void k_edge2_a(const int* __restrict__ ei, const float* __restrict__ as2,
                          const float* __restrict__ ad2, float* __restrict__ e2,
                          unsigned* __restrict__ mx2) {
  long e = (long)blockIdx.x * blockDim.x + threadIdx.x;
  if (e >= kET) return;
  int s = (e < kE) ? ei[e] : (int)(e - kE);
  int d = (e < kE) ? ei[kE + e] : (int)(e - kE);
  float v = as2[s] + ad2[d];
  v = (v > 0.f) ? v : 0.2f * v;
  e2[e] = v;
  atomicMax(&mx2[d], f2ord(v));
}

__global__ void k_edge2_b(const int* __restrict__ ei, float* __restrict__ e2,
                          const unsigned* __restrict__ mx2, float* __restrict__ den2) {
  long e = (long)blockIdx.x * blockDim.x + threadIdx.x;
  if (e >= kET) return;
  int d = (e < kE) ? ei[kE + e] : (int)(e - kE);
  float ex = expf(e2[e] - ord2f(mx2[d]));
  e2[e] = ex;
  atomicAdd(&den2[d], ex);
}

__global__ void k_edge2_c(const int* __restrict__ ei, const float* __restrict__ ex2,
                          const float* __restrict__ den2, const float* __restrict__ hh,
                          float* __restrict__ out2, float* __restrict__ alpha_out) {
  long e = (long)blockIdx.x * blockDim.x + threadIdx.x;
  if (e >= kET) return;
  int s = (e < kE) ? ei[e] : (int)(e - kE);
  int d = (e < kE) ? ei[kE + e] : (int)(e - kE);
  float a = ex2[e] / den2[d];
  alpha_out[e] = a;                            // alpha2 output
  const float4* __restrict__ hs = reinterpret_cast<const float4*>(hh + (size_t)s * kOUT);
  float* __restrict__ od = out2 + (size_t)d * kOUT;
  #pragma unroll
  for (int q = 0; q < 10; q++) {               // 10 float4 = 40 channels
    float4 m = hs[q];
    atomicAdd(&od[q * 4 + 0], a * m.x);
    atomicAdd(&od[q * 4 + 1], a * m.y);
    atomicAdd(&od[q * 4 + 2], a * m.z);
    atomicAdd(&od[q * 4 + 3], a * m.w);
  }
}

// ---------- finalize: +b2, raw out (in place) and log_softmax ----------
__global__ void k_finalize(float* __restrict__ out2, const float* __restrict__ b2,
                           float* __restrict__ logsm) {
  int n = blockIdx.x * blockDim.x + threadIdx.x;
  if (n >= kN) return;
  float v[kOUT];
  float m = -3.4e38f;
  float* row = out2 + (size_t)n * kOUT;
  #pragma unroll 8
  for (int j = 0; j < kOUT; j++) {
    v[j] = row[j] + b2[j];
    m = fmaxf(m, v[j]);
  }
  float sum = 0.f;
  #pragma unroll 8
  for (int j = 0; j < kOUT; j++) sum += expf(v[j] - m);
  float lse = logf(sum) + m;
  float* lrow = logsm + (size_t)n * kOUT;
  #pragma unroll 8
  for (int j = 0; j < kOUT; j++) {
    row[j] = v[j];
    lrow[j] = v[j] - lse;
  }
}

// =====================================================================

static inline size_t align256(size_t x) { return (x + 255) & ~(size_t)255; }

extern "C" void kernel_launch(void* const* d_in, const int* in_sizes, int n_in,
                              void* d_out, int out_size, void* d_ws, size_t ws_size,
                              hipStream_t stream) {
  const float* x     = (const float*)d_in[0];
  const int*   ei    = (const int*)d_in[1];
  const float* W1    = (const float*)d_in[2];
  const float* asrc1 = (const float*)d_in[3];
  const float* adst1 = (const float*)d_in[4];
  const float* b1    = (const float*)d_in[5];
  const float* W2    = (const float*)d_in[6];
  const float* asrc2 = (const float*)d_in[7];
  const float* adst2 = (const float*)d_in[8];
  const float* b2    = (const float*)d_in[9];

  // output layout: [log_softmax N*40][alpha2 ET][out N*40]
  float* o_logsm = (float*)d_out;
  float* o_alpha = o_logsm + (size_t)kN * kOUT;
  float* o_out2  = o_alpha + (size_t)kET;

  // workspace carve-out
  char* ws = (char*)d_ws;
  size_t off = 0;
  auto carve = [&](size_t bytes) { size_t r = off; off += align256(bytes); return r; };
  _Float16* xh   = (_Float16*)(ws + carve((size_t)kN * kIN * 2));
  _Float16* w1t  = (_Float16*)(ws + carve((size_t)kH1 * kIN * 2));
  float*    h1   = (float*)(ws + carve((size_t)kN * kH1 * 4));
  float*    as1  = (float*)(ws + carve((size_t)kN * kHEADS * 4));
  float*    ad1  = (float*)(ws + carve((size_t)kN * kHEADS * 4));
  unsigned* mx1  = (unsigned*)(ws + carve((size_t)kN * kHEADS * 4));
  float*    den1 = (float*)(ws + carve((size_t)kN * kHEADS * 4));
  float*    e1   = (float*)(ws + carve((size_t)kET * kHEADS * 4));
  float*    out1 = (float*)(ws + carve((size_t)kN * kH1 * 4));   // becomes h2 after ELU
  float*    hh   = (float*)(ws + carve((size_t)kN * kOUT * 4));
  float*    as2  = (float*)(ws + carve((size_t)kN * 4));
  float*    ad2  = (float*)(ws + carve((size_t)kN * 4));
  unsigned* mx2  = (unsigned*)(ws + carve((size_t)kN * 4));
  float*    den2 = (float*)(ws + carve((size_t)kN * 4));
  float*    e2   = (float*)(ws + carve((size_t)kET * 4));

  const int T = 256;
  auto blk = [](long n, int t) { return (unsigned)((n + t - 1) / t); };

  // convert inputs to f16 (x resident in L2 afterwards: 102.4 MB < 192 MB)
  k_cvt_f16<<<blk((long)kN * kIN, T), T, 0, stream>>>(x, xh, (long)kN * kIN);
  k_w1_transpose<<<blk(kH1 * kIN, T), T, 0, stream>>>(W1, w1t);

  // GEMM1 (f16 WMMA)
  k_gemm1<<<3125, 256, 0, stream>>>(xh, w1t, h1);

  // layer-1 attention
  k_nodevec1<<<blk((long)kN * kHEADS, T), T, 0, stream>>>(h1, asrc1, adst1, as1, ad1);
  k_fill_u32<<<blk((long)kN * kHEADS, T), T, 0, stream>>>(mx1, 0u, (long)kN * kHEADS);
  k_fill_u32<<<blk((long)kN * kHEADS, T), T, 0, stream>>>((unsigned*)den1, 0u, (long)kN * kHEADS);
  k_fill_u32<<<blk((long)kN * kH1, T), T, 0, stream>>>((unsigned*)out1, 0u, (long)kN * kH1);

  unsigned eblk = blk(kET, T);
  k_edge1_a<<<eblk, T, 0, stream>>>(ei, as1, ad1, e1, mx1);
  k_edge1_b<<<eblk, T, 0, stream>>>(ei, e1, mx1, den1);
  k_edge1_c<<<eblk, T, 0, stream>>>(ei, e1, den1, h1, out1);

  // ELU(+b1) -> h2 (in place)
  k_elu<<<blk((long)kN * kH1, T), T, 0, stream>>>(out1, b1, (long)kN * kH1);

  // GEMM2 (f32 WMMA)
  k_gemm2<<<2344, 256, 0, stream>>>(out1, W2, hh);

  // layer-2 attention
  k_nodevec2<<<blk(kN, T), T, 0, stream>>>(hh, asrc2, adst2, as2, ad2);
  k_fill_u32<<<blk(kN, T), T, 0, stream>>>(mx2, 0u, kN);
  k_fill_u32<<<blk(kN, T), T, 0, stream>>>((unsigned*)den2, 0u, kN);
  k_fill_u32<<<blk((long)kN * kOUT, T), T, 0, stream>>>((unsigned*)o_out2, 0u, (long)kN * kOUT);

  k_edge2_a<<<eblk, T, 0, stream>>>(ei, as2, ad2, e2, mx2);
  k_edge2_b<<<eblk, T, 0, stream>>>(ei, e2, mx2, den2);
  k_edge2_c<<<eblk, T, 0, stream>>>(ei, e2, den2, hh, o_out2, o_alpha);

  // bias + log_softmax
  k_finalize<<<blk(kN, T), T, 0, stream>>>(o_out2, b2, o_logsm);
}
